// LIFConv2dLayer_39548058862036
// MI455X (gfx1250) — matrix operand order repaired and checked by reading, not compiled, not run on hardware
//
#include <hip/hip_runtime.h>

typedef __attribute__((ext_vector_type(16))) _Float16 v16h;
typedef __attribute__((ext_vector_type(8)))  float    v8f;
typedef __attribute__((ext_vector_type(4)))  int      v4i;

namespace {
constexpr int BB = 64, CC = 64, HH = 64, WW = 64;
constexpr int HW    = HH * WW;              // 4096
constexpr int CHW   = CC * HW;              // 262144
constexpr int NTENS = BB * CHW;             // 16777216
constexpr int PHH = HH / 2, PWW = WW / 2;   // 32, 32
constexpr int NPOOL = BB * CC * PHH * PWW;  // 4194304
constexpr int NOUT  = 10;
constexpr int NFLAT = CC * PHH * PWW;       // 65536

// output layout: (pooled_S, rreadout, argmax, P_new, Q_new, R_out, U) flat-concatenated
constexpr long long OFF_POOLED = 0;
constexpr long long OFF_RREAD  = OFF_POOLED + (long long)NPOOL;
constexpr long long OFF_ARGMAX = OFF_RREAD + (long long)BB * NOUT;
constexpr long long OFF_P      = OFF_ARGMAX + BB;
constexpr long long OFF_Q      = OFF_P + (long long)NTENS;
constexpr long long OFF_R      = OFF_Q + (long long)NTENS;
constexpr long long OFF_U      = OFF_R + (long long)NTENS;

constexpr int KSTEPS = 18;                        // (kh,kw) taps x 2 ci-halves of 32
constexpr int AFRAG_HALFS = 4 * KSTEPS * 32 * 16; // 36864 halves = 72KB
constexpr int HBLK  = 4;                          // output rows per workgroup
constexpr int LROWS = HBLK + 2;                   // staged rows (with halo)
constexpr int LDS_HALFS = LROWS * 66 * 64;        // 25344 halves = 50.7KB

// workspace layout
constexpr size_t WS_AFRAG = 0;
constexpr size_t WS_UAUX  = WS_AFRAG + (size_t)AFRAG_HALFS * 2;          // 73728
constexpr size_t WS_P16   = WS_UAUX + (size_t)NPOOL * 4;                 // +16MB
}

// -------- CDNA5 async global->LDS copy (guarded; falls back to ds_store) ----
#if defined(__gfx1250__) && __has_builtin(__builtin_amdgcn_global_load_async_to_lds_b128)
#define HAVE_ASYNC_LDS 1
#else
#define HAVE_ASYNC_LDS 0
#endif

#if HAVE_ASYNC_LDS
typedef v4i __attribute__((address_space(1))) as1v4i;
typedef v4i __attribute__((address_space(3))) as3v4i;
#define ASYNC_CP16(gsrc, ldst)                                                 \
  __builtin_amdgcn_global_load_async_to_lds_b128(                              \
      (as1v4i*)(void*)(gsrc), (as3v4i*)(void*)(ldst), 0, 0)

__device__ __forceinline__ void async_wait0() {
#if __has_builtin(__builtin_amdgcn_s_wait_asynccnt)
  __builtin_amdgcn_s_wait_asynccnt(0);
#else
  asm volatile("s_wait_asynccnt 0" ::: "memory");
#endif
}
#endif

// ---------------------------------------------------------------------------
// Kernel 0: permute weights (f32 -> f16) into per-lane WMMA A-fragment layout.
// A is 16x32 (MxK) f16; lane m = L%16, group g = L/16;
// VGPR j<4 holds K = 2j+half + 8g, VGPR j>=4 holds K = 16+2(j-4)+half + 8g.
// ---------------------------------------------------------------------------
__global__ void prep_afrag_kernel(const float* __restrict__ wts,
                                  _Float16* __restrict__ afrag) {
  int t = blockIdx.x * 256 + threadIdx.x;
  if (t >= 4 * KSTEPS * 32) return;
  int mtile = t / (KSTEPS * 32);
  int rest  = t % (KSTEPS * 32);
  int ks    = rest >> 5;
  int lane  = rest & 31;
  int g = lane >> 4, m = lane & 15;
  int co = mtile * 16 + m;
  int pos = ks >> 1, cih = ks & 1;
  int kh = pos / 3, kw = pos % 3;
#pragma unroll
  for (int i = 0; i < 16; ++i) {
    int j = i >> 1, hlf = i & 1;
    int kl = (j < 4 ? 2 * j + hlf : 16 + 2 * (j - 4) + hlf) + g * 8;
    int ci = cih * 32 + kl;
    float wv = wts[((co * 64 + ci) * 3 + kh) * 3 + kw];
    afrag[(long long)t * 16 + i] = (_Float16)wv;
  }
}

// ---------------------------------------------------------------------------
// Kernel 1: fused LIF state update + transposed f16 emission.
// One block per (b,h): reads P,Q,input coalesced, writes P_new/Q_new (f32),
// and a ci-contiguous f16 copy P16[b][h][w][ci] via an LDS transpose so the
// global write is fully coalesced 16B chunks.
// ---------------------------------------------------------------------------
__global__ __launch_bounds__(256)
void state_kernel(const float* __restrict__ in_t, const float* __restrict__ P,
                  const float* __restrict__ Q, float* __restrict__ Pn,
                  float* __restrict__ Qn, _Float16* __restrict__ P16) {
  __shared__ alignas(128) _Float16 tsh[64 * 64];  // [w][ci]
  const int b = blockIdx.x >> 6;
  const int h = blockIdx.x & 63;
  for (int idx = threadIdx.x; idx < CC * WW; idx += 256) {
    int ci = idx >> 6, w = idx & 63;
    long long g = (long long)b * CHW + ci * HW + h * WW + w;
    float p = P[g], q = Q[g], x = in_t[g];
    float pn = 0.95f * p + 20.0f * q;
    Pn[g] = pn;
    Qn[g] = 0.87f * q + 7.5f * x;
    tsh[w * 64 + ci] = (_Float16)pn;
  }
  __syncthreads();
  _Float16* dst = P16 + ((long long)(b * 64 + h)) * (WW * CC);
  for (int c = threadIdx.x; c < 512; c += 256)     // 512 x 16B = 8KB row
    ((uint4*)dst)[c] = ((const uint4*)tsh)[c];
}

// ---------------------------------------------------------------------------
// Kernel 2: 3x3 conv as implicit GEMM on WMMA (f16 in, f32 acc).
// One workgroup per (b, 4-row h-block): stage 6 rows of P16 (already f16,
// [w][ci] ci-contiguous) into LDS with async global->LDS b128 copies, then
// 8 waves x 2 tiles x 4 rows of 16x16x(576) WMMA GEMM. Epilogue fuses
// R_new = 0.65*R, U, spike threshold, R_out.
// ---------------------------------------------------------------------------
__global__ __launch_bounds__(256)
void conv_wmma_kernel(const _Float16* __restrict__ P16, const float* __restrict__ R,
                      const _Float16* __restrict__ afrag,
                      float* __restrict__ U, float* __restrict__ Rout) {
  __shared__ alignas(128) _Float16 lds[LDS_HALFS];  // [row][w+halo][ci]
  const int tid = threadIdx.x;
  const int b  = blockIdx.x >> 4;
  const int h0 = (blockIdx.x & 15) * HBLK;

  // zero w-halo columns (w=-1 and w=64) for all staged rows
  for (int i = tid; i < LROWS * 64; i += 256) {
    int r = i >> 6, ci = i & 63;
    lds[(r * 66 + 0)  * 64 + ci] = (_Float16)0.0f;
    lds[(r * 66 + 65) * 64 + ci] = (_Float16)0.0f;
  }
  // zero out-of-image rows (only at top/bottom image borders)
  for (int r = 0; r < LROWS; ++r) {
    int hr = h0 - 1 + r;
    if (hr >= 0 && hr < HH) continue;
    for (int i = tid; i < 66 * 64; i += 256) lds[r * 66 * 64 + i] = (_Float16)0.0f;
  }
  // stage valid rows: pure contiguous 16B-chunk copies (async-to-LDS path)
  for (int r = 0; r < LROWS; ++r) {
    int hr = h0 - 1 + r;
    if (hr < 0 || hr >= HH) continue;
    const _Float16* src = P16 + ((long long)(b * 64 + hr)) * (WW * CC);
    for (int c = tid; c < 512; c += 256) {
      int w = c >> 3, ci8 = c & 7;
      _Float16* dst = &lds[(r * 66 + (w + 1)) * 64 + ci8 * 8];
#if HAVE_ASYNC_LDS
      ASYNC_CP16(src + c * 8, dst);
#else
      *(uint4*)dst = *(const uint4*)(src + c * 8);
#endif
    }
  }
#if HAVE_ASYNC_LDS
  async_wait0();
#endif
  __syncthreads();

  const int wave = tid >> 5, lane = tid & 31;
  const int g = lane >> 4, n = lane & 15;

  for (int hl = 0; hl < HBLK; ++hl) {
    const int h = h0 + hl;
    for (int t = 0; t < 2; ++t) {
      const int tile  = wave + t * 8;   // 0..15
      const int mtile = tile & 3;       // Cout block of 16
      const int w0    = (tile >> 2) * 16;
      v8f acc = {};
#pragma unroll
      for (int ks = 0; ks < KSTEPS; ++ks) {
        const int pos = ks >> 1, cih = ks & 1;
        const int kh = pos / 3, kw = pos % 3;
        v16h a  = *(const v16h*)(afrag + ((mtile * KSTEPS + ks) * 32 + lane) * 16);
        const _Float16* bp = lds + ((hl + kh) * 66 + (w0 + n + kw)) * 64 +
                                 cih * 32 + g * 8;
        v16h bm;
#pragma unroll
        for (int s = 0; s < 16; ++s) bm[s] = bp[s < 8 ? s : s + 8];
        acc = __builtin_amdgcn_wmma_f32_16x16x32_f16(false, a, false, bm,
                                                     (short)0, acc, false, false);
      }
      // D layout: VGPR vv holds M = vv + 8*g, N = lane%16
      const int w = w0 + n;
#pragma unroll
      for (int vv = 0; vv < 8; ++vv) {
        int co = mtile * 16 + vv + 8 * g;
        long long gidx = (long long)b * CHW + co * HW + h * WW + w;
        float rnew = 0.65f * R[gidx];
        float u = acc[vv] + rnew;
        float s = (u >= 1.0f) ? 1.0f : 0.0f;
        U[gidx] = u;
        Rout[gidx] = rnew - s;
      }
    }
  }
}

// ---------------------------------------------------------------------------
// Kernel 3: fused pooling. max over 2x2 of U; pooled_S = (maxU >= 1),
// U_aux = sigmoid(maxU) (sigmoid monotonic => commutes with max).
// ---------------------------------------------------------------------------
__global__ __launch_bounds__(256)
void pool_kernel(const float* __restrict__ U, float* __restrict__ pooledS,
                 float* __restrict__ uaux) {
  int idx = blockIdx.x * 256 + threadIdx.x;  // [B][C][PH][PW] flat
  int pw = idx & 31, ph = (idx >> 5) & 31;
  int c = (idx >> 10) & 63, b = idx >> 16;
  long long base = (long long)b * CHW + c * HW + (2 * ph) * WW + 2 * pw;
  float u00 = U[base],      u01 = U[base + 1];
  float u10 = U[base + WW], u11 = U[base + WW + 1];
  float m = fmaxf(fmaxf(u00, u01), fmaxf(u10, u11));
  pooledS[idx] = (m >= 1.0f) ? 1.0f : 0.0f;
  uaux[idx] = 1.0f / (1.0f + __expf(-m));
}

// ---------------------------------------------------------------------------
// Kernel 4: readout GEMV [B x 65536] . [10 x 65536]^T + argmax.
// ---------------------------------------------------------------------------
__global__ __launch_bounds__(256)
void readout_kernel(const float* __restrict__ uaux, const float* __restrict__ rw,
                    float* __restrict__ rread, float* __restrict__ amax) {
  __shared__ float red[256];
  __shared__ float rr[NOUT];
  int b = blockIdx.x, tid = threadIdx.x;
  float acc[NOUT];
#pragma unroll
  for (int o = 0; o < NOUT; ++o) acc[o] = 0.0f;
  for (int f = tid; f < NFLAT; f += 256) {
    float ua = uaux[(long long)b * NFLAT + f];
#pragma unroll
    for (int o = 0; o < NOUT; ++o)
      acc[o] = fmaf(ua, rw[(long long)o * NFLAT + f], acc[o]);
  }
  for (int o = 0; o < NOUT; ++o) {
    red[tid] = acc[o];
    __syncthreads();
    for (int s = 128; s > 0; s >>= 1) {
      if (tid < s) red[tid] += red[tid + s];
      __syncthreads();
    }
    if (tid == 0) rr[o] = red[0];
    __syncthreads();
  }
  if (tid == 0) {
    int best = 0;
    for (int o = 0; o < NOUT; ++o) {
      rread[b * NOUT + o] = rr[o];
      if (rr[o] > rr[best]) best = o;   // strict >: first-max, matches jnp.argmax
    }
    amax[b] = (float)best;
  }
}

// ---------------------------------------------------------------------------
extern "C" void kernel_launch(void* const* d_in, const int* in_sizes, int n_in,
                              void* d_out, int out_size, void* d_ws, size_t ws_size,
                              hipStream_t stream) {
  const float* in_t = (const float*)d_in[0];
  const float* P    = (const float*)d_in[1];
  const float* Q    = (const float*)d_in[2];
  const float* R    = (const float*)d_in[3];
  const float* wts  = (const float*)d_in[4];
  const float* rw   = (const float*)d_in[5];
  // d_in[6] = y_local (unused by the returned outputs)

  float*    out   = (float*)d_out;
  _Float16* afrag = (_Float16*)((char*)d_ws + WS_AFRAG);
  float*    uaux  = (float*)((char*)d_ws + WS_UAUX);
  _Float16* p16   = (_Float16*)((char*)d_ws + WS_P16);

  prep_afrag_kernel<<<(4 * KSTEPS * 32 + 255) / 256, 256, 0, stream>>>(wts, afrag);
  state_kernel<<<BB * HH, 256, 0, stream>>>(in_t, P, Q, out + OFF_P, out + OFF_Q, p16);
  conv_wmma_kernel<<<BB * (HH / HBLK), 256, 0, stream>>>(p16, R, afrag,
                                                         out + OFF_U, out + OFF_R);
  pool_kernel<<<NPOOL / 256, 256, 0, stream>>>(out + OFF_U, out + OFF_POOLED, uaux);
  readout_kernel<<<BB, 256, 0, stream>>>(uaux, rw, out + OFF_RREAD, out + OFF_ARGMAX);
}
